// EncoderLayer_25744033973123
// MI455X (gfx1250) — matrix operand order repaired and checked by reading, not compiled, not run on hardware
//
#include <hip/hip_runtime.h>
#include <math.h>

typedef unsigned short u16;
typedef __attribute__((ext_vector_type(16))) __bf16 v16bf;
typedef __attribute__((ext_vector_type(8)))  __bf16 v8bf;
typedef __attribute__((ext_vector_type(8)))  float  v8f;

#define D_MODEL 1024
#define N_HEAD  16
#define DKH     64
#define D_FF    2048
#define BATCH   16
#define SEQ     512
#define MTOT    (BATCH*SEQ)   // 8192

#define WMMA_BF16(A_, B_, C_) \
  __builtin_amdgcn_wmma_f32_16x16x32_bf16(false, (A_), false, (B_), (short)0, (C_), false, false)

__device__ __forceinline__ u16 f2bf(float f) {
  unsigned u = __float_as_uint(f);
  u += 0x7FFFu + ((u >> 16) & 1u);       // round-to-nearest-even
  return (u16)(u >> 16);
}

__device__ __forceinline__ v16bf combine16(v8bf lo, v8bf hi) {
  v16bf r;
#pragma unroll
  for (int i = 0; i < 8; i++) { r[i] = lo[i]; r[i + 8] = hi[i]; }
  return r;
}

__device__ __forceinline__ v8f vzero8() {
  v8f z = {0.f, 0.f, 0.f, 0.f, 0.f, 0.f, 0.f, 0.f};
  return z;
}

// CDNA5 async data mover: global -> LDS, 16B per lane, tracked by ASYNCcnt.
// LDS address = low 32 bits of the generic pointer (aperture rule addr[31:0]).
__device__ __forceinline__ void async_b128(const u16* gptr, u16* lptr) {
  unsigned      lds = (unsigned)(unsigned long long)lptr;
  unsigned long long ga = (unsigned long long)gptr;
  asm volatile("global_load_async_to_lds_b128 %0, %1, off"
               :: "v"(lds), "v"(ga) : "memory");
}
#define WAIT_ASYNC(n) asm volatile("s_wait_asynccnt " #n ::: "memory")

// ---------------- fp32 -> bf16 elementwise ----------------
__global__ __launch_bounds__(256)
void cvt_bf16_kernel(const float* __restrict__ in, u16* __restrict__ out) {
  size_t i = ((size_t)blockIdx.x * 256 + threadIdx.x) * 4;
  float4 v = *(const float4*)(in + i);
  unsigned lo = (unsigned)f2bf(v.x) | ((unsigned)f2bf(v.y) << 16);
  unsigned hi = (unsigned)f2bf(v.z) | ((unsigned)f2bf(v.w) << 16);
  *(uint2*)(out + i) = make_uint2(lo, hi);
}

// ------------- W[K x N] fp32 -> WT[N x K] bf16 -------------
__global__ __launch_bounds__(256)
void transpose_bf16_kernel(const float* __restrict__ in, u16* __restrict__ out,
                           int K, int N) {
  __shared__ float tile[32][33];
  int k0 = blockIdx.y * 32, n0 = blockIdx.x * 32;
  int tx = threadIdx.x, ty = threadIdx.y;   // 32 x 8
#pragma unroll
  for (int i = 0; i < 32; i += 8)
    tile[ty + i][tx] = in[(size_t)(k0 + ty + i) * N + n0 + tx];
  __syncthreads();
#pragma unroll
  for (int i = 0; i < 32; i += 8)
    out[(size_t)(n0 + ty + i) * K + k0 + tx] = f2bf(tile[tx][ty + i]);
}

// --------------------- generic WMMA GEMM ---------------------
// C[M x N] = A[M x K](bf16) @ B, with B supplied transposed BT[N x K](bf16).
// Double-buffered LDS, tiles streamed by the async data mover (ASYNCcnt).
// MODE 0: Cb = bf16(C + bias)
// MODE 1: V-mode: scatter bf16(C + bias) into Vt[B,H,DK,S]
// MODE 2: Cf = C + bias + resid (fp32)
// MODE 3: Cb = bf16(gelu_exact(C + bias))
template<int MODE>
__global__ __launch_bounds__(256)
void gemm_bf16_kernel(const u16* __restrict__ A, const u16* __restrict__ BT,
                      const float* __restrict__ bias, const float* __restrict__ resid,
                      float* __restrict__ Cf, u16* __restrict__ Cb,
                      int M, int N, int K) {
  __shared__ __align__(32) u16 sA[2][128 * 32];
  __shared__ __align__(32) u16 sB[2][128 * 32];
  const int m0 = blockIdx.y * 128, n0 = blockIdx.x * 128;
  const int t = threadIdx.x, lane = t & 31, wave = t >> 5;
  const int wm = (wave >> 2) * 64, wn = (wave & 3) * 32;
  const int ln = lane & 15, hlf = lane >> 4;
  const int lrow = t >> 1, lseg = (t & 1) * 16;
  const int koff = hlf * 8, kb = hlf * 16;

  const u16* gA = A  + (size_t)(m0 + lrow) * K + lseg;
  const u16* gB = BT + (size_t)(n0 + lrow) * K + lseg;
  u16* lA = &sA[0][lrow * 32 + lseg];
  u16* lB = &sB[0][lrow * 32 + lseg];

  v8f acc[4][2];
#pragma unroll
  for (int i = 0; i < 4; i++)
#pragma unroll
    for (int j = 0; j < 2; j++) acc[i][j] = vzero8();

  // prologue: stream tile 0 into buffer 0
  async_b128(gA,     lA);
  async_b128(gA + 8, lA + 8);
  async_b128(gB,     lB);
  async_b128(gB + 8, lB + 8);

  for (int k0 = 0; k0 < K; k0 += 32) {
    const int cur = (k0 >> 5) & 1;
    if (k0 + 32 < K) {
      // stream next tile into the other buffer while we compute on this one
      const int nb = (cur ^ 1) * (128 * 32);
      async_b128(gA + k0 + 32,     lA + nb);
      async_b128(gA + k0 + 40,     lA + nb + 8);
      async_b128(gB + k0 + 32,     lB + nb);
      async_b128(gB + k0 + 40,     lB + nb + 8);
      if (k0 + 64 < K) {           // warm GL2 one tile deeper
        __builtin_prefetch(gA + k0 + 64, 0, 0);
        __builtin_prefetch(gB + k0 + 64, 0, 0);
      }
      WAIT_ASYNC(0x4);             // current tile done; next 4 still in flight
    } else {
      WAIT_ASYNC(0x0);
    }
    __syncthreads();

    v16bf af[4];
#pragma unroll
    for (int mt = 0; mt < 4; mt++) {
      int row = wm + mt * 16 + ln;
      v8bf lo = *(const v8bf*)&sA[cur][row * 32 + koff];
      v8bf hi = *(const v8bf*)&sA[cur][row * 32 + 16 + koff];
      af[mt] = combine16(lo, hi);
    }
    v16bf bfv[2];
#pragma unroll
    for (int nt = 0; nt < 2; nt++) {
      int n = wn + nt * 16 + ln;
      bfv[nt] = combine16(*(const v8bf*)&sB[cur][n * 32 + koff], *(const v8bf*)&sB[cur][n * 32 + 16 + koff]);
    }
#pragma unroll
    for (int mt = 0; mt < 4; mt++)
#pragma unroll
      for (int nt = 0; nt < 2; nt++)
        acc[mt][nt] = WMMA_BF16(af[mt], bfv[nt], acc[mt][nt]);
    __syncthreads();               // all waves done reading before buffer reuse
  }

#pragma unroll
  for (int mt = 0; mt < 4; mt++) {
#pragma unroll
    for (int nt = 0; nt < 2; nt++) {
      int col = n0 + wn + nt * 16 + ln;
      float bval = bias[col];
#pragma unroll
      for (int r = 0; r < 8; r++) {
        int row = m0 + wm + mt * 16 + r + hlf * 8;
        float v = acc[mt][nt][r] + bval;
        size_t idx = (size_t)row * N + col;
        if (MODE == 0) {
          Cb[idx] = f2bf(v);
        } else if (MODE == 1) {
          int bb = row >> 9, ss = row & 511, hh = col >> 6, dk = col & 63;
          Cb[(((size_t)(bb * N_HEAD + hh) * DKH + dk) << 9) + ss] = f2bf(v);
        } else if (MODE == 2) {
          Cf[idx] = v + resid[idx];
        } else {
          float g = 0.5f * v * (1.0f + erff(v * 0.70710678118654752f));
          Cb[idx] = f2bf(g);
        }
      }
    }
  }
}

// --------------------- fused attention ---------------------
// grid (SEQ/128, N_HEAD, BATCH); 256 threads. Flash-style online softmax.
__global__ __launch_bounds__(256)
void attention_kernel(const u16* __restrict__ Qb, const u16* __restrict__ Kb,
                      const u16* __restrict__ Vtb, const int* __restrict__ mask,
                      u16* __restrict__ Ob) {
  __shared__ __align__(32) u16 sK[128 * 64];   // K tile [sk][dk]  16KB
  __shared__ __align__(32) u16 sV[64 * 128];   // Vt tile [dk][sk] 16KB
  __shared__ __align__(32) u16 sP[128 * 128];  // probs bf16       32KB
  const int b = blockIdx.z, h = blockIdx.y, q0 = blockIdx.x * 128;
  const int t = threadIdx.x, lane = t & 31, wave = t >> 5;
  const int ln = lane & 15, hlf = lane >> 4;
  const int koff = hlf * 8, kb = hlf * 16;

  // Q fragments kept in registers: this wave's 16 rows, DK=64 -> 2 k-steps.
  v16bf qf[2];
  {
    const u16* qrow = Qb + (size_t)(b * SEQ + q0 + wave * 16 + ln) * D_MODEL + h * DKH;
#pragma unroll
    for (int ks = 0; ks < 2; ks++) {
      v8bf lo = *(const v8bf*)(qrow + ks * 32 + koff);
      v8bf hi = *(const v8bf*)(qrow + ks * 32 + 16 + koff);
      qf[ks] = combine16(lo, hi);
    }
  }

  // per-thread async-copy source/dest for K and V tiles
  const int krow = t >> 1, kseg = (t & 1) * 32;   // 2 thr/row, 64B each
  const int vrow = t >> 2, vseg = (t & 3) * 32;   // 4 thr/row, 64B each
  const u16* gK = Kb  + (size_t)(b * SEQ + krow) * D_MODEL + h * DKH + kseg;
  const u16* gV = Vtb + ((size_t)(b * N_HEAD + h) * DKH + vrow) * SEQ + vseg;
  u16* lK = &sK[krow * 64 + kseg];
  u16* lV = &sV[vrow * 128 + vseg];

  float mrow[8], lrow[8];
  v8f oacc[4];
#pragma unroll
  for (int r = 0; r < 8; r++) { mrow[r] = -1e30f; lrow[r] = 0.f; }
#pragma unroll
  for (int nt = 0; nt < 4; nt++) oacc[nt] = vzero8();

  for (int sk0 = 0; sk0 < SEQ; sk0 += 128) {
    __syncthreads();                 // previous tile fully consumed
    // stream K tile (4 x b128) and Vt tile (4 x b128) via the data mover
    {
      const u16* gk = gK + (size_t)sk0 * D_MODEL;
      async_b128(gk,      lK);      async_b128(gk + 8,  lK + 8);
      async_b128(gk + 16, lK + 16); async_b128(gk + 24, lK + 24);
      const u16* gv = gV + sk0;
      async_b128(gv,      lV);      async_b128(gv + 8,  lV + 8);
      async_b128(gv + 16, lV + 16); async_b128(gv + 24, lV + 24);
    }
    WAIT_ASYNC(0x0);
    __syncthreads();

    // scores S = Q @ K^T : 16 x 128 per wave
    v8f sacc[8];
#pragma unroll
    for (int nt = 0; nt < 8; nt++) sacc[nt] = vzero8();
#pragma unroll
    for (int ks = 0; ks < 2; ks++) {
#pragma unroll
      for (int nt = 0; nt < 8; nt++) {
        v16bf bfv = combine16(*(const v8bf*)&sK[(nt * 16 + ln) * 64 + ks * 32 + koff], *(const v8bf*)&sK[(nt * 16 + ln) * 64 + ks * 32 + 16 + koff]);
        sacc[nt] = WMMA_BF16(qf[ks], bfv, sacc[nt]);
      }
    }

    // mask + scale, running row max
    float mnew[8], fac[8];
#pragma unroll
    for (int r = 0; r < 8; r++) {
      float m = -1e30f;
#pragma unroll
      for (int nt = 0; nt < 8; nt++) {
        int col = nt * 16 + ln;
        float sv = mask[b * SEQ + sk0 + col] ? sacc[nt][r] * 0.125f : -1e10f;
        sacc[nt][r] = sv;
        m = fmaxf(m, sv);
      }
#pragma unroll
      for (int o = 8; o > 0; o >>= 1) m = fmaxf(m, __shfl_xor(m, o, 32));
      mnew[r] = fmaxf(mrow[r], m);
      fac[r] = __expf(mrow[r] - mnew[r]);
      mrow[r] = mnew[r];
    }

    // probs -> bf16 LDS, row sums, rescale running state
#pragma unroll
    for (int r = 0; r < 8; r++) {
      float rs = 0.f;
#pragma unroll
      for (int nt = 0; nt < 8; nt++) {
        float p = __expf(sacc[nt][r] - mnew[r]);
        rs += p;
        sP[(wave * 16 + r + hlf * 8) * 128 + nt * 16 + ln] = f2bf(p);
      }
#pragma unroll
      for (int o = 8; o > 0; o >>= 1) rs += __shfl_xor(rs, o, 32);
      lrow[r] = lrow[r] * fac[r] + rs;
    }
#pragma unroll
    for (int nt = 0; nt < 4; nt++)
#pragma unroll
      for (int r = 0; r < 8; r++) oacc[nt][r] *= fac[r];

    // O += P @ V  (each wave reads only its own 16 P rows; per-wave LDS order ok)
#pragma unroll
    for (int ks2 = 0; ks2 < 4; ks2++) {
      int arow = wave * 16 + ln;
      v8bf lo = *(const v8bf*)&sP[arow * 128 + ks2 * 32 + koff];
      v8bf hi = *(const v8bf*)&sP[arow * 128 + ks2 * 32 + 16 + koff];
      v16bf pf = combine16(lo, hi);
#pragma unroll
      for (int nt = 0; nt < 4; nt++) {
        v16bf vf = combine16(*(const v8bf*)&sV[(nt * 16 + ln) * 128 + ks2 * 32 + koff], *(const v8bf*)&sV[(nt * 16 + ln) * 128 + ks2 * 32 + 16 + koff]);
        oacc[nt] = WMMA_BF16(pf, vf, oacc[nt]);
      }
    }
  }

  // finalize: divide by row sums, emit bf16 [B,S,D]
#pragma unroll
  for (int nt = 0; nt < 4; nt++) {
#pragma unroll
    for (int r = 0; r < 8; r++) {
      int row = q0 + wave * 16 + r + hlf * 8;
      int col = h * DKH + nt * 16 + ln;
      float o = oacc[nt][r] / lrow[r];
      Ob[(size_t)(b * SEQ + row) * D_MODEL + col] = f2bf(o);
    }
  }
}

// --------------------- LayerNorm (torch-style) ---------------------
// one block (256 thr) per row of 1024; unbiased variance, eps added to std.
__global__ __launch_bounds__(256)
void layernorm_kernel(const float* __restrict__ in, const float* __restrict__ gamma,
                      const float* __restrict__ beta, float* __restrict__ outf,
                      u16* __restrict__ outb) {
  const int row = blockIdx.x, t = threadIdx.x, lane = t & 31, wave = t >> 5;
  const float* x = in + (size_t)row * D_MODEL;
  float4 v = *(const float4*)(x + t * 4);
  float s = v.x + v.y + v.z + v.w;
#pragma unroll
  for (int o = 16; o > 0; o >>= 1) s += __shfl_xor(s, o, 32);
  __shared__ float red[8], red2[8];
  if (lane == 0) red[wave] = s;
  __syncthreads();
  float tot = 0.f;
#pragma unroll
  for (int i = 0; i < 8; i++) tot += red[i];
  float mean = tot * (1.0f / D_MODEL);
  float d0 = v.x - mean, d1 = v.y - mean, d2 = v.z - mean, d3 = v.w - mean;
  float sq = d0 * d0 + d1 * d1 + d2 * d2 + d3 * d3;
#pragma unroll
  for (int o = 16; o > 0; o >>= 1) sq += __shfl_xor(sq, o, 32);
  if (lane == 0) red2[wave] = sq;
  __syncthreads();
  float sqtot = 0.f;
#pragma unroll
  for (int i = 0; i < 8; i++) sqtot += red2[i];
  float stdv = sqrtf(sqtot * (1.0f / (D_MODEL - 1)));
  float inv = 1.0f / (stdv + 1e-6f);
  float4 g = *(const float4*)(gamma + t * 4);
  float4 be = *(const float4*)(beta + t * 4);
  float y0 = g.x * d0 * inv + be.x, y1 = g.y * d1 * inv + be.y;
  float y2 = g.z * d2 * inv + be.z, y3 = g.w * d3 * inv + be.w;
  float* of = outf + (size_t)row * D_MODEL + t * 4;
  *(float4*)of = make_float4(y0, y1, y2, y3);
  if (outb) {
    unsigned lo = (unsigned)f2bf(y0) | ((unsigned)f2bf(y1) << 16);
    unsigned hi = (unsigned)f2bf(y2) | ((unsigned)f2bf(y3) << 16);
    *(uint2*)(outb + (size_t)row * D_MODEL + t * 4) = make_uint2(lo, hi);
  }
}

// =====================================================================
extern "C" void kernel_launch(void* const* d_in, const int* in_sizes, int n_in,
                              void* d_out, int out_size, void* d_ws, size_t ws_size,
                              hipStream_t stream) {
  (void)in_sizes; (void)n_in; (void)out_size; (void)ws_size;
  const float* x   = (const float*)d_in[0];
  const int* xmask = (const int*)  d_in[1];
  const float* Wq = (const float*)d_in[2];  const float* bq = (const float*)d_in[3];
  const float* Wk = (const float*)d_in[4];  const float* bk = (const float*)d_in[5];
  const float* Wv = (const float*)d_in[6];  const float* bv = (const float*)d_in[7];
  const float* Wo = (const float*)d_in[8];  const float* bo = (const float*)d_in[9];
  const float* g1 = (const float*)d_in[10]; const float* be1 = (const float*)d_in[11];
  const float* W1 = (const float*)d_in[12]; const float* b1 = (const float*)d_in[13];
  const float* W2 = (const float*)d_in[14]; const float* b2 = (const float*)d_in[15];
  const float* g2 = (const float*)d_in[16]; const float* be2 = (const float*)d_in[17];
  float* out = (float*)d_out;

  char* ws = (char*)d_ws;
  size_t o = 0;
  auto take = [&](size_t bytes) -> size_t {
    size_t p = o; o += (bytes + 255) & ~(size_t)255; return p;
  };
  const size_t MD2 = (size_t)MTOT * D_MODEL * 2;
  u16*   xb   = (u16*)(ws + take(MD2));
  u16*   WqT  = (u16*)(ws + take((size_t)D_MODEL * D_MODEL * 2));
  u16*   WkT  = (u16*)(ws + take((size_t)D_MODEL * D_MODEL * 2));
  u16*   WvT  = (u16*)(ws + take((size_t)D_MODEL * D_MODEL * 2));
  u16*   WoT  = (u16*)(ws + take((size_t)D_MODEL * D_MODEL * 2));
  u16*   W1T  = (u16*)(ws + take((size_t)D_FF * D_MODEL * 2));   // [2048][1024]
  u16*   W2T  = (u16*)(ws + take((size_t)D_MODEL * D_FF * 2));   // [1024][2048]
  u16*   Qb   = (u16*)(ws + take(MD2));
  u16*   Kb   = (u16*)(ws + take(MD2));
  u16*   Vtb  = (u16*)(ws + take(MD2));
  u16*   Ob   = (u16*)(ws + take(MD2));
  float* ares = (float*)(ws + take((size_t)MTOT * D_MODEL * 4));
  float* hbuf = (float*)(ws + take((size_t)MTOT * D_MODEL * 4));
  u16*   hb   = xb;            // reuse: xb dead after V projection
  u16*   fb   = Qb;            // reuse: Qb+Kb contiguous 32MB, dead after attention
  float* tmp2 = ares;          // reuse: ares dead after LN1

  dim3 blk(256);
  // 1) activations & weights -> bf16 (weights transposed)
  cvt_bf16_kernel<<<(MTOT * D_MODEL) / 1024, blk, 0, stream>>>(x, xb);
  dim3 tb(32, 8);
  transpose_bf16_kernel<<<dim3(D_MODEL / 32, D_MODEL / 32), tb, 0, stream>>>(Wq, WqT, D_MODEL, D_MODEL);
  transpose_bf16_kernel<<<dim3(D_MODEL / 32, D_MODEL / 32), tb, 0, stream>>>(Wk, WkT, D_MODEL, D_MODEL);
  transpose_bf16_kernel<<<dim3(D_MODEL / 32, D_MODEL / 32), tb, 0, stream>>>(Wv, WvT, D_MODEL, D_MODEL);
  transpose_bf16_kernel<<<dim3(D_MODEL / 32, D_MODEL / 32), tb, 0, stream>>>(Wo, WoT, D_MODEL, D_MODEL);
  transpose_bf16_kernel<<<dim3(D_FF / 32, D_MODEL / 32), tb, 0, stream>>>(W1, W1T, D_MODEL, D_FF);
  transpose_bf16_kernel<<<dim3(D_MODEL / 32, D_FF / 32), tb, 0, stream>>>(W2, W2T, D_FF, D_MODEL);

  // 2) QKV projections (WMMA, async-DMA tiles)
  dim3 gQ(D_MODEL / 128, MTOT / 128);
  gemm_bf16_kernel<0><<<gQ, blk, 0, stream>>>(xb, WqT, bq, nullptr, nullptr, Qb, MTOT, D_MODEL, D_MODEL);
  gemm_bf16_kernel<0><<<gQ, blk, 0, stream>>>(xb, WkT, bk, nullptr, nullptr, Kb, MTOT, D_MODEL, D_MODEL);
  gemm_bf16_kernel<1><<<gQ, blk, 0, stream>>>(xb, WvT, bv, nullptr, nullptr, Vtb, MTOT, D_MODEL, D_MODEL);

  // 3) fused attention (WMMA flash-style)
  attention_kernel<<<dim3(SEQ / 128, N_HEAD, BATCH), blk, 0, stream>>>(Qb, Kb, Vtb, xmask, Ob);

  // 4) output projection + residual, LN1
  gemm_bf16_kernel<2><<<gQ, blk, 0, stream>>>(Ob, WoT, bo, x, ares, nullptr, MTOT, D_MODEL, D_MODEL);
  layernorm_kernel<<<MTOT, blk, 0, stream>>>(ares, g1, be1, hbuf, hb);

  // 5) FFN: GELU GEMM then down-proj + residual, LN2
  dim3 gF1(D_FF / 128, MTOT / 128);
  gemm_bf16_kernel<3><<<gF1, blk, 0, stream>>>(hb, W1T, b1, nullptr, nullptr, fb, MTOT, D_FF, D_MODEL);
  gemm_bf16_kernel<2><<<gQ, blk, 0, stream>>>(fb, W2T, b2, hbuf, tmp2, nullptr, MTOT, D_MODEL, D_FF);
  layernorm_kernel<<<MTOT, blk, 0, stream>>>(tmp2, g2, be2, out, nullptr);
}
